// LocalWindowTripleLoss_64604898066562
// MI455X (gfx1250) — matrix operand order repaired and checked by reading, not compiled, not run on hardware
//
#include <hip/hip_runtime.h>

// ---- problem constants (from reference) ----
#define B_     16
#define C_     32
#define H_     176
#define MH     44      // map spatial
#define KW     11      // unfold kernel
#define TAPS   121     // 11*11
#define STR    5
#define OW     34      // (176-11)/5 + 1
#define NWIN   (OW*OW)
#define ALPHA  10.0f
#define MARGIN 0.5f
#define EPSV   1e-6f

typedef __attribute__((ext_vector_type(16))) _Float16 v16h;
typedef __attribute__((ext_vector_type(8)))  float    v8f;

union V16H { v16h v; _Float16 h[16]; };
union V8F  { v8f  v; float     f[8];  };

// full-wave (wave32) sum reduction; every lane gets the result
__device__ __forceinline__ float wred32(float v) {
#pragma unroll
  for (int m = 16; m; m >>= 1) v += __shfl_xor(v, m, 32);
  return v;
}

// Low 32 bits of a flat shared-aperture pointer are the LDS byte offset
// (ISA 10.2).
__device__ __forceinline__ unsigned lds_off_of(const void* p) {
  return (unsigned)(unsigned long long)(uintptr_t)p;
}

__global__ void init_ws_kernel(float* ws) {
  if (threadIdx.x < 2 * B_) ws[threadIdx.x] = 0.0f;
}

// One block = one (batch, window). 128 threads = 4 waves.
__global__ void __launch_bounds__(128)
win_kernel(const float* __restrict__ mp, const float* __restrict__ feat,
           float* __restrict__ wsS, float* __restrict__ wsV) {
  __shared__ float s_fu[128][33];   // [tap][channel], rows 121..127 zeroed
  __shared__ float s_us[128];       // soft map per tap (padded 0)
  __shared__ float s_dotp[128];     // fu . pc  per tap (WMMA output)
  __shared__ float s_dotn[128];     // fu . nc  per tap (WMMA output)
  __shared__ float s_red0[4][32];
  __shared__ float s_red1[4][32];
  __shared__ float s_pc[32], s_nc[32];
  __shared__ float s_pos, s_validf, s_pp2, s_nn2;

  const int tid = threadIdx.x;
  const int b   = blockIdx.x / NWIN;
  const int w   = blockIdx.x % NWIN;
  const int wy  = (w / OW) * STR;
  const int wx  = (w % OW) * STR;

  // ---- issue async feat-tile fill: global -> LDS transpose scatter -------
  // s_fu[tap][c] = feat[b][c][wy+ty][wx+tx]; tracked by ASYNCcnt.
  if (tid < TAPS) {
    const int ty = tid / KW, tx = tid % KW;
    const float* gp = feat + (((size_t)b * C_) * H_ + (wy + ty)) * H_ + (wx + tx);
    const unsigned lo = lds_off_of(&s_fu[tid][0]);
#pragma unroll
    for (int c = 0; c < C_; ++c) {
      asm volatile("global_load_async_to_lds_b32 %0, %1, off"
                   :: "v"(lo + 4u * (unsigned)c),
                      "v"(gp + (size_t)c * H_ * H_)
                   : "memory");
    }
  } else {
    // zero the pad rows so the WMMA A-tiles see zeros
#pragma unroll
    for (int c = 0; c < 33; ++c) s_fu[tid][c] = 0.0f;
  }

  // ---- softmap taps: fused bilinear 4x upsample (overlaps async fill) ----
  {
    float us = 0.0f;
    if (tid < TAPS) {
      const int y = wy + tid / KW;
      const int x = wx + tid % KW;
      const float sy = 0.25f * (float)y - 0.375f;   // (y+0.5)*0.25 - 0.5
      const float sx = 0.25f * (float)x - 0.375f;
      const float fy = floorf(sy), fx = floorf(sx);
      const float ay = sy - fy,   ax = sx - fx;
      int y0 = (int)fy, x0 = (int)fx;
      const int y1 = min(max(y0 + 1, 0), MH - 1);
      const int x1 = min(max(x0 + 1, 0), MH - 1);
      y0 = min(max(y0, 0), MH - 1);
      x0 = min(max(x0, 0), MH - 1);
      const float* mb = mp + (size_t)b * MH * MH;
      const float m00 = mb[y0 * MH + x0], m01 = mb[y0 * MH + x1];
      const float m10 = mb[y1 * MH + x0], m11 = mb[y1 * MH + x1];
      us = (1.f - ay) * ((1.f - ax) * m00 + ax * m01) +
                  ay  * ((1.f - ax) * m10 + ax * m11);
    }
    s_us[tid] = us;
  }

  asm volatile("s_wait_asynccnt 0" ::: "memory");
  __syncthreads();

  // ---- wave 0: soft pos count + hard validity ----
  if (tid < 32) {
    float su = 0.f, hp = 0.f;
#pragma unroll
    for (int k = 0; k < 4; ++k) {
      const float u = s_us[tid + 32 * k];   // pad taps are exactly 0
      su += u;
      hp += (u > 0.5f) ? 1.f : 0.f;
    }
    su = wred32(su);
    hp = wred32(hp);
    if (tid == 0) {
      s_pos    = su;
      s_validf = (hp > 0.5f && ((float)TAPS - hp) > 0.5f) ? 1.f : 0.f;
    }
  }

  // ---- all threads: partial centroid sums (c = tid%32, 4-way tap split) ----
  {
    const int c = tid & 31, part = tid >> 5;
    float a0 = 0.f, a1 = 0.f;
    for (int t = part; t < TAPS; t += 4) {
      const float f = s_fu[t][c];
      a0 += f;
      a1 += s_us[t] * f;
    }
    s_red0[part][c] = a0;
    s_red1[part][c] = a1;
  }
  __syncthreads();

  // ---- wave 0: centroids pc/nc and their squared norms ----
  if (tid < 32) {
    const int c = tid;
    const float s0 = s_red0[0][c] + s_red0[1][c] + s_red0[2][c] + s_red0[3][c];
    const float s1 = s_red1[0][c] + s_red1[1][c] + s_red1[2][c] + s_red1[3][c];
    const float pos = s_pos;
    const float neg = (float)TAPS - pos;
    const float pc = s1 / (pos + EPSV);
    const float nc = (s0 - s1) / (neg + EPSV);
    s_pc[c] = pc;
    s_nc[c] = nc;
    const float pp2 = wred32(pc * pc);
    const float nn2 = wred32(nc * nc);
    if (tid == 0) { s_pp2 = pp2; s_nn2 = nn2; }
  }
  __syncthreads();

  // ---- wave 0: WMMA  D[121x2] = Fu[121x32(f16)] x [pc|nc][32x2(f16)] ----
  // V_WMMA_F32_16X16X32_F16, 8 tap-tiles, K=32 == channel count.
  if (tid < 32) {
    const int lane = tid;
    const int n  = lane & 15;   // C/D column, B column, A row
    const int hi = lane >> 4;

    // B (32x16 f16): lane=N; halves i<8 hold K = i + 8*hi, i>=8 hold K = 16 + 8*hi + (i-8). col0=pc, col1=nc.
    V16H bv;
#pragma unroll
    for (int i = 0; i < 16; ++i) {
      const int k = (i < 8) ? (i + (hi ? 8 : 0)) : (16 + (hi ? 8 : 0) + (i - 8));
      const float v = (n == 0) ? s_pc[k] : (n == 1) ? s_nc[k] : 0.0f;
      bv.h[i] = (_Float16)v;
    }

#pragma unroll
    for (int tile = 0; tile < 8; ++tile) {
      const int tap0 = tile * 16;
      // A (16x32 f16): lane%16 = row M; VGPR v<4: K=2v+h+8*hi, v>=4: K=16+2(v-4)+h+8*hi
      V16H av;
#pragma unroll
      for (int i = 0; i < 16; ++i) {
        const int vv = i >> 1, hh = i & 1;
        const int k = (vv < 4) ? (2 * vv + hh + (hi ? 8 : 0))
                               : (16 + 2 * (vv - 4) + hh + (hi ? 8 : 0));
        av.h[i] = (_Float16)s_fu[tap0 + n][k];
      }
      V8F d;
#pragma unroll
      for (int r = 0; r < 8; ++r) d.f[r] = 0.0f;

      d.v = __builtin_amdgcn_wmma_f32_16x16x32_f16(
          /*neg_a=*/false, av.v, /*neg_b=*/false, bv.v,
          /*c_mod=*/(short)0, d.v, /*reuse_a=*/false, /*reuse_b=*/false);

      // D(M,N): VGPR=M%8, lane=N+16*(M>=8). Only cols 0 (pc) / 1 (nc) used.
      // Lanes 0,1,16,17  <=>  (lane & 14) == 0.
      if ((lane & 14) == 0) {
        float* dst = (n == 0) ? s_dotp : s_dotn;
        const int o = tap0 + hi * 8;
#pragma unroll
        for (int r = 0; r < 8; ++r) dst[o + r] = d.f[r];
      }
    }
  }
  __syncthreads();

  // ---- per-tap distances, triple margin, window accumulation ----
  float tri = 0.0f;
  if (tid < TAPS) {
    float f2 = 0.0f;
#pragma unroll
    for (int c = 0; c < C_; ++c) { const float f = s_fu[tid][c]; f2 += f * f; }
    const float qp = f2 - 2.0f * s_dotp[tid] + s_pp2;  // ||fu - pc||^2
    const float qn = f2 - 2.0f * s_dotn[tid] + s_nn2;  // ||fu - nc||^2
    const float dp = 1.0f - __expf(-ALPHA * qp);
    const float dn = 1.0f - __expf(-ALPHA * qn);
    const bool isp = s_us[tid] > 0.5f;
    const float pd = isp ? dp : dn;
    const float nd = isp ? dn : dp;
    tri = fmaxf(pd - nd + MARGIN, 0.0f);
  }
  __syncthreads();          // s_dotp reads done before scratch reuse
  s_dotp[tid] = tri;        // reuse as reduction scratch
  __syncthreads();

  if (tid < 32) {
    float s = s_dotp[tid] + s_dotp[tid + 32] + s_dotp[tid + 64] + s_dotp[tid + 96];
    s = wred32(s);
    if (tid == 0) {
      const float vf = s_validf;
      atomicAdd(&wsS[b], s * vf);
      atomicAdd(&wsV[b], vf);
    }
  }
}

__global__ void final_kernel(const float* __restrict__ ws, float* __restrict__ out) {
  const int t = threadIdx.x;
  float v = 0.0f;
  if (t < B_) v = ws[t] / (ws[B_ + t] + EPSV);
  v = wred32(v);
  if (t == 0) out[0] = v / (float)(B_ * TAPS);
}

extern "C" void kernel_launch(void* const* d_in, const int* in_sizes, int n_in,
                              void* d_out, int out_size, void* d_ws, size_t ws_size,
                              hipStream_t stream) {
  (void)in_sizes; (void)n_in; (void)out_size; (void)ws_size;
  const float* mp   = (const float*)d_in[0];  // [16,1,44,44]
  const float* feat = (const float*)d_in[1];  // [16,32,176,176]
  float* ws  = (float*)d_ws;                  // [0..15]=S_b, [16..31]=V_b
  float* out = (float*)d_out;                 // scalar

  init_ws_kernel<<<1, 32, 0, stream>>>(ws);
  win_kernel<<<B_ * NWIN, 128, 0, stream>>>(mp, feat, ws, ws + B_);
  final_kernel<<<1, 32, 0, stream>>>(ws, out);
}
